// DPCABlock_39676907888286
// MI455X (gfx1250) — matrix operand order repaired and checked by reading, not compiled, not run on hardware
//
#include <hip/hip_runtime.h>
#include <hip/hip_bf16.h>

// ---------------------------------------------------------------------------
// CDNA5 / gfx1250 implementation of the DPCA block.
// All GEMMs use V_WMMA_F32_16X16X32_BF16 (wave32 WMMA), bf16 in / f32 acc.
// Round 4: fix async builtin pointer types — the toolchain's
// __builtin_amdgcn_global_load_async_to_lds_b128 takes (v4i AS1*, v4i AS3*,
// imm offset, imm cpol) per the round-3 diagnostic.
// ---------------------------------------------------------------------------

typedef __bf16 bf16_t;
typedef __attribute__((ext_vector_type(16))) __bf16 v16bf;
typedef __attribute__((ext_vector_type(8)))  float  v8f;
typedef __attribute__((__vector_size__(4 * sizeof(int)))) int v4i_t;

#if defined(__AMDGCN__) && \
    __has_builtin(__builtin_amdgcn_global_load_async_to_lds_b128) && \
    __has_builtin(__builtin_amdgcn_s_wait_asynccnt)
#define USE_ASYNC_LDS 1
#else
#define USE_ASYNC_LDS 0
#endif

constexpr int  Bc    = 4;
constexpr int  CHc   = 256;
constexpr int  Lc    = 4096;     // 64*64
constexpr int  HEADSc= 8;
constexpr int  DHc   = 32;
constexpr int  MLPc  = 1024;
constexpr int  BHc   = Bc * HEADSc;    // 32
constexpr int  TOPKc = 513;            // 1 + L/HEADS
constexpr int  TKPc  = 576;            // padded to 9*64: tiles both N and K paths
constexpr float EPSc = 1e-5f;

// ---------------------------------------------------------------------------
// WMMA fragment helpers — layouts per CDNA5 ISA §7.12.2 (wave32).
// ---------------------------------------------------------------------------

union frag_u { int4 i[2]; v16bf v; };

// A fragment from row-major LDS tile (ld elements per row, 16B-aligned runs).
// lane: m = lane&15, kb = (lane>=16)*8; elems 0..7 -> K=kb+e, 8..15 -> 16+kb+e.
__device__ __forceinline__ v16bf load_a_frag(const bf16_t* a, int lda) {
  int lane = threadIdx.x & 31;
  int m  = lane & 15;
  int kb = (lane >> 4) << 3;
  const bf16_t* row = a + m * lda;
  frag_u u;
  u.i[0] = *(const int4*)(row + kb);
  u.i[1] = *(const int4*)(row + 16 + kb);
  return u.v;
}

// B fragment from TRANSPOSED LDS tile Bs[n][k] (ldb elements per n-row).
// lane: n = lane&15, kb = (lane>=16)*8; elems 0..7 -> K=kb+e, 8..15 -> 16+kb+e.
__device__ __forceinline__ v16bf load_b_fragT(const bf16_t* bt, int ldb) {
  int lane = threadIdx.x & 31;
  const bf16_t* p = bt + (lane & 15) * ldb + ((lane >> 4) << 3);
  frag_u u;
  u.i[0] = *(const int4*)(p);
  u.i[1] = *(const int4*)(p + 16);
  return u.v;
}

// C/D 16x16 f32: vgpr r -> M = 8*(lane>=16)+r, N = lane&15.
__device__ __forceinline__ void store_c_frag(float* C, int ldc, const v8f& acc) {
  int lane = threadIdx.x & 31;
  int n  = lane & 15;
  int mh = (lane >> 4) << 3;
#pragma unroll
  for (int r = 0; r < 8; ++r) C[(long)(mh + r) * ldc + n] = acc[r];
}

// ---------------------------------------------------------------------------
// Tiled bf16 GEMM: C[M,N] = A[M,K] * B[K,N]; A row-major (lda=K), B row-major
// (ldb=N), C f32 (ldc=N). Requires M%64==0, N%NT==0, K%32==0.
// Block = 128 threads = 4 waves; tile 64 x NT; LDS double-buffered; B tile
// stored transposed with padded stride so fragments are pure ds_load_b128.
// A tile staged with async load-to-LDS (ASYNCcnt) when available.
// ---------------------------------------------------------------------------
template <int NT>
__global__ __launch_bounds__(128)
void gemm_bf16_kernel(const bf16_t* __restrict__ A, long strideA,
                      const bf16_t* __restrict__ B, long strideB,
                      float* __restrict__ C, long strideC,
                      int M, int N, int K) {
  constexpr int NT16 = NT / 16;
  constexpr int LDA  = 40;   // 80 B row stride: 16B-aligned, bank-skewed
  constexpr int LDB  = 40;

  __shared__ __align__(16) bf16_t As[2][64 * LDA];
  __shared__ __align__(16) bf16_t Bs[2][NT * LDB];

  const int batch = blockIdx.z;
  A += (long)batch * strideA;
  B += (long)batch * strideB;
  C += (long)batch * strideC;

  const int m0   = blockIdx.y * 64;
  const int n0   = blockIdx.x * NT;
  const int tid  = threadIdx.x;
  const int wave = tid >> 5;

  auto stage = [&](int kt, int buf) {
    const int k0 = kt * 32;
    // ---- A tile 64x32: row-major copy, b128 chunks (2 per thread). ----
#if USE_ASYNC_LDS
#pragma unroll
    for (int c = tid; c < 256; c += 128) {
      int r = c >> 2, seg = (c & 3) * 8;
      __builtin_amdgcn_global_load_async_to_lds_b128(
          (__attribute__((address_space(1))) v4i_t*)
              (A + (long)(m0 + r) * K + k0 + seg),
          (__attribute__((address_space(3))) v4i_t*)&As[buf][r * LDA + seg],
          0, 0);
    }
#else
#pragma unroll
    for (int c = tid; c < 256; c += 128) {
      int r = c >> 2, seg = (c & 3) * 8;
      *(int4*)&As[buf][r * LDA + seg] =
          *(const int4*)(A + (long)(m0 + r) * K + k0 + seg);
    }
#endif
    // ---- B tile 32xNT: load phase (overlapped), then transposed scatter. ----
    int4 bv[NT / 32];
#pragma unroll
    for (int i = 0; i < NT / 32; ++i) {
      int c = tid + i * 128;
      int r = c / (NT / 8), seg = (c % (NT / 8)) * 8;
      bv[i] = *(const int4*)(B + (long)(k0 + r) * N + n0 + seg);
    }
#pragma unroll
    for (int i = 0; i < NT / 32; ++i) {
      int c = tid + i * 128;
      int r = c / (NT / 8), seg = (c % (NT / 8)) * 8;
      const bf16_t* pv = (const bf16_t*)&bv[i];
#pragma unroll
      for (int e = 0; e < 8; ++e)
        Bs[buf][(seg + e) * LDB + r] = pv[e];
    }
  };

  v8f acc[NT16] = {};

  const int nk = K >> 5;
  stage(0, 0);
#if USE_ASYNC_LDS
  __builtin_amdgcn_s_wait_asynccnt(0);   // barriers do NOT cover ASYNCcnt
#endif
  __syncthreads();

  for (int kt = 0; kt < nk; ++kt) {
    const int buf = kt & 1;
    if (kt + 1 < nk) stage(kt + 1, buf ^ 1);     // overlap with WMMA below
    if (kt + 2 < nk) {                           // global_prefetch_b8 k+2
      __builtin_prefetch(A + (long)(m0 + (tid >> 2)) * K + (kt + 2) * 32, 0, 1);
      __builtin_prefetch(B + (long)((kt + 2) * 32 + (tid >> 3)) * N + n0, 0, 1);
    }

    v16bf af = load_a_frag(&As[buf][(wave * 16) * LDA], LDA);
#pragma unroll
    for (int nf = 0; nf < NT16; ++nf) {
      v16bf bf = load_b_fragT(&Bs[buf][(nf * 16) * LDB], LDB);
      acc[nf] = __builtin_amdgcn_wmma_f32_16x16x32_bf16(
          false, af, false, bf, (short)0, acc[nf], false, false);
    }
#if USE_ASYNC_LDS
    __builtin_amdgcn_s_wait_asynccnt(0);
#endif
    __syncthreads();
  }

  float* Cw = C + (long)(m0 + wave * 16) * N + n0;
#pragma unroll
  for (int nf = 0; nf < NT16; ++nf) store_c_frag(Cw + nf * 16, N, acc[nf]);
}

// ---------------------------------------------------------------------------
// Elementwise / reduction kernels
// ---------------------------------------------------------------------------

__global__ __launch_bounds__(256)
void f2bf_kernel(const float* __restrict__ s, bf16_t* __restrict__ d, long n) {
  long t = (long)blockIdx.x * 256 + threadIdx.x;
  if (t < n) d[t] = (bf16_t)s[t];
}

// Channel LayerNorm over C=256 for each (b,l); one block per column.
__global__ __launch_bounds__(256)
void chan_ln_kernel(const float* __restrict__ x, const float* __restrict__ g,
                    const float* __restrict__ bp, bf16_t* __restrict__ ob,
                    float* __restrict__ of) {
  int bl = blockIdx.x;
  int b  = bl / Lc, l = bl - b * Lc;
  int c  = threadIdx.x;
  long base = ((long)b * CHc + c) * Lc + l;
  float v = x[base];
  __shared__ float s1[256], s2[256];
  s1[c] = v; s2[c] = v * v;
  __syncthreads();
  for (int o = 128; o; o >>= 1) {
    if (c < o) { s1[c] += s1[c + o]; s2[c] += s2[c + o]; }
    __syncthreads();
  }
  float mu  = s1[0] * (1.0f / CHc);
  float var = s2[0] * (1.0f / CHc) - mu * mu;
  float y = (v - mu) * rsqrtf(var + EPSc) * g[c] + bp[c];
  ob[base] = (bf16_t)y;
  if (of) of[base] = y;
}

// LN(out)*g+b, then r = gamma*ln + x (residual), emit f32 + bf16.
__global__ __launch_bounds__(256)
void ln_axpy_kernel(const float* __restrict__ p, const float* __restrict__ g,
                    const float* __restrict__ bp, const float* __restrict__ gamma,
                    const float* __restrict__ xres,
                    float* __restrict__ aof, bf16_t* __restrict__ aob) {
  int bl = blockIdx.x;
  int b  = bl / Lc, l = bl - b * Lc;
  int c  = threadIdx.x;
  long base = ((long)b * CHc + c) * Lc + l;
  float v = p[base];
  __shared__ float s1[256], s2[256];
  s1[c] = v; s2[c] = v * v;
  __syncthreads();
  for (int o = 128; o; o >>= 1) {
    if (c < o) { s1[c] += s1[c + o]; s2[c] += s2[c + o]; }
    __syncthreads();
  }
  float mu  = s1[0] * (1.0f / CHc);
  float var = s2[0] * (1.0f / CHc) - mu * mu;
  float y = (v - mu) * rsqrtf(var + EPSc) * g[c] + bp[c];
  float r = gamma[0] * y + xres[base];
  aof[base] = r;
  aob[base] = (bf16_t)r;
}

// Fold (B,C,L)->(BH,L,DH) and l2-normalize q and k rows; q also to bf16.
__global__ __launch_bounds__(256)
void fold_l2_kernel(const float* __restrict__ qraw, const float* __restrict__ kvraw,
                    float* __restrict__ qf, bf16_t* __restrict__ qb,
                    float* __restrict__ kf) {
  long t = (long)blockIdx.x * 256 + threadIdx.x;
  if (t >= (long)BHc * Lc) return;
  int l  = (int)(t % Lc);
  int bh = (int)(t / Lc);
  int b = bh / HEADSc, h = bh - b * HEADSc;
  float qv[DHc], kv[DHc];
  float qs = 0.f, ks = 0.f;
#pragma unroll
  for (int d = 0; d < DHc; ++d) {
    qv[d] = qraw[((long)b * CHc + h * DHc + d) * Lc + l];
    kv[d] = kvraw[((long)b * 2 * CHc + h * DHc + d) * Lc + l];
    qs += qv[d] * qv[d];
    ks += kv[d] * kv[d];
  }
  float qi = 1.0f / fmaxf(sqrtf(qs), 1e-12f);
  float ki = 1.0f / fmaxf(sqrtf(ks), 1e-12f);
  long base = t * DHc;
#pragma unroll
  for (int d = 0; d < DHc; ++d) {
    float qn = qv[d] * qi;
    qf[base + d] = qn;
    qb[base + d] = (bf16_t)qn;
    kf[base + d] = kv[d] * ki;
  }
}

// Gather Q_small rows at rand_ind.
__global__ __launch_bounds__(256)
void gather_qs_kernel(const float* __restrict__ qf, const int* __restrict__ rind,
                      float* __restrict__ Qs) {
  long t = (long)blockIdx.x * 256 + threadIdx.x;
  if (t >= (long)BHc * TOPKc * DHc) return;
  int d  = (int)(t & (DHc - 1));
  long r = t >> 5;
  int j  = (int)(r % TOPKc);
  int bh = (int)(r / TOPKc);
  int l  = rind[(long)bh * TOPKc + j];
  Qs[t] = qf[((long)bh * Lc + l) * DHc + d];
}

// min over 513 sampled queries of L1 distance to every key row.
__global__ __launch_bounds__(256)
void l1_min_kernel(const float* __restrict__ kf, const float* __restrict__ Qs,
                   float* __restrict__ min_d) {
  int bh = blockIdx.y;
  int l  = blockIdx.x * 256 + threadIdx.x;
  float kr[DHc];
  const float* krow = kf + ((long)bh * Lc + l) * DHc;
#pragma unroll
  for (int d = 0; d < DHc; ++d) kr[d] = krow[d];
  __shared__ float qs[32 * DHc];
  const float* Qb = Qs + (long)bh * TOPKc * DHc;
  float best = __builtin_inff();
  for (int j0 = 0; j0 < TOPKc; j0 += 32) {
    int cnt = (TOPKc - j0 < 32) ? (TOPKc - j0) : 32;
    int tot = cnt * DHc;
    for (int i = threadIdx.x; i < tot; i += 256) qs[i] = Qb[j0 * DHc + i];
    __syncthreads();
    for (int j = 0; j < cnt; ++j) {
      float s = 0.f;
#pragma unroll
      for (int d = 0; d < DHc; ++d) s += fabsf(kr[d] - qs[j * DHc + d]);
      best = fminf(best, s);
    }
    __syncthreads();
  }
  min_d[(long)bh * Lc + l] = best;
}

// Deterministic top-k (smallest) selection by rank counting (stable ties).
__global__ __launch_bounds__(256)
void select_topk_kernel(const float* __restrict__ min_d, int* __restrict__ idx) {
  int bh = blockIdx.x;
  __shared__ float d[Lc];
  const float* src = min_d + (long)bh * Lc;
  for (int i = threadIdx.x; i < Lc; i += 256) d[i] = src[i];
  __syncthreads();
  for (int l = threadIdx.x; l < Lc; l += 256) {
    float dl = d[l];
    int rank = 0;
    for (int j = 0; j < Lc; ++j) {
      float dj = d[j];
      rank += (dj < dl) || (dj == dl && j < l);
    }
    if (rank < TOPKc) idx[(long)bh * TKPc + rank] = l;
  }
}

// Gather k_sel^T (BH,DH,TKP) and v_sel (BH,TKP,DH) as bf16; zero pad rows.
__global__ __launch_bounds__(256)
void gather_kv_kernel(const int* __restrict__ idx, const float* __restrict__ kf,
                      const float* __restrict__ kvf,
                      bf16_t* __restrict__ kselT, bf16_t* __restrict__ vsel) {
  long t = (long)blockIdx.x * 256 + threadIdx.x;
  if (t >= (long)BHc * TKPc * DHc) return;
  int d  = (int)(t & (DHc - 1));
  long r = t >> 5;
  int j  = (int)(r % TKPc);
  int bh = (int)(r / TKPc);
  int b = bh / HEADSc, h = bh - b * HEADSc;
  float kk = 0.f, vv = 0.f;
  if (j < TOPKc) {
    int l = idx[(long)bh * TKPc + j];
    kk = kf[((long)bh * Lc + l) * DHc + d];
    vv = kvf[((long)b * 2 * CHc + CHc + h * DHc + d) * Lc + l];
  }
  kselT[((long)bh * DHc + d) * TKPc + j] = (bf16_t)kk;
  vsel[((long)bh * TKPc + j) * DHc + d]  = (bf16_t)vv;
}

// Row softmax over 513 valid of 576 columns; one wave (32 lanes) per row.
__global__ __launch_bounds__(256)
void softmax_kernel(const float* __restrict__ S, bf16_t* __restrict__ A) {
  constexpr int NJ = TKPc / 32;   // 18
  int row  = blockIdx.x * 8 + (threadIdx.x >> 5);
  int lane = threadIdx.x & 31;
  const float* s = S + (long)row * TKPc;
  float vals[NJ];
  float mx = -__builtin_inff();
#pragma unroll
  for (int i = 0; i < NJ; ++i) {
    int j = lane + i * 32;
    float v = (j < TOPKc) ? s[j] : -__builtin_inff();
    vals[i] = v;
    mx = fmaxf(mx, v);
  }
#pragma unroll
  for (int o = 16; o; o >>= 1) mx = fmaxf(mx, __shfl_xor(mx, o, 32));
  float sum = 0.f;
#pragma unroll
  for (int i = 0; i < NJ; ++i) {
    int j = lane + i * 32;
    float e = (j < TOPKc) ? __expf(vals[i] - mx) : 0.f;
    vals[i] = e;
    sum += e;
  }
#pragma unroll
  for (int o = 16; o; o >>= 1) sum += __shfl_xor(sum, o, 32);
  float inv = 1.0f / sum;
#pragma unroll
  for (int i = 0; i < NJ; ++i) {
    int j = lane + i * 32;
    A[(long)row * TKPc + j] = (bf16_t)(vals[i] * inv);
  }
}

// (BH,L,DH) f32 -> (B,C,L) bf16 un-fold for the W_out projection input.
__global__ __launch_bounds__(256)
void unfold_kernel(const float* __restrict__ out4, bf16_t* __restrict__ dst) {
  long t = (long)blockIdx.x * 256 + threadIdx.x;
  if (t >= (long)Bc * CHc * Lc) return;
  int l  = (int)(t % Lc);
  long r = t / Lc;
  int c  = (int)(r % CHc);
  int b  = (int)(r / CHc);
  int h = c >> 5, d = c & 31;
  int bh = b * HEADSc + h;
  dst[t] = (bf16_t)out4[((long)bh * Lc + l) * DHc + d];
}

// MLP stage 1 epilogue: bf16(relu(bn1_g*(h+b1)+bn1_b)).
__global__ __launch_bounds__(256)
void relu_bn_kernel(const float* __restrict__ h, const float* __restrict__ b1,
                    const float* __restrict__ g1, const float* __restrict__ s1,
                    bf16_t* __restrict__ hb) {
  long t = (long)blockIdx.x * 256 + threadIdx.x;
  if (t >= (long)Bc * MLPc * Lc) return;
  int m = (int)((t / Lc) % MLPc);
  float v = g1[m] * (h[t] + b1[m]) + s1[m];
  hb[t] = (bf16_t)fmaxf(v, 0.f);
}

// Final residual: y = attn_out + bn2_g*(h2+b2)+bn2_b.
__global__ __launch_bounds__(256)
void final_kernel(const float* __restrict__ ao, const float* __restrict__ h2,
                  const float* __restrict__ b2, const float* __restrict__ g2,
                  const float* __restrict__ s2, float* __restrict__ y) {
  long t = (long)blockIdx.x * 256 + threadIdx.x;
  if (t >= (long)Bc * CHc * Lc) return;
  int c = (int)((t / Lc) % CHc);
  y[t] = ao[t] + g2[c] * (h2[t] + b2[c]) + s2[c];
}

// ---------------------------------------------------------------------------
// Host side
// ---------------------------------------------------------------------------

static inline void launch_gemm(const bf16_t* A, long sA, const bf16_t* B, long sB,
                               float* C, long sC, int M, int N, int K, int batch,
                               hipStream_t st) {
  if (N % 64 == 0) {
    dim3 g(N / 64, M / 64, batch), b(128);
    gemm_bf16_kernel<64><<<g, b, 0, st>>>(A, sA, B, sB, C, sC, M, N, K);
  } else {
    dim3 g(N / 32, M / 64, batch), b(128);
    gemm_bf16_kernel<32><<<g, b, 0, st>>>(A, sA, B, sB, C, sC, M, N, K);
  }
}

extern "C" void kernel_launch(void* const* d_in, const int* in_sizes, int n_in,
                              void* d_out, int out_size, void* d_ws, size_t ws_size,
                              hipStream_t stream) {
  (void)in_sizes; (void)n_in; (void)out_size; (void)ws_size;
  const float* qsrc  = (const float*)d_in[0];
  const float* ctx   = (const float*)d_in[1];
  const int*   rind  = (const int*)d_in[2];
  const float* cn_g  = (const float*)d_in[3];
  const float* cn_b  = (const float*)d_in[4];
  const float* qn_g  = (const float*)d_in[5];
  const float* qn_b  = (const float*)d_in[6];
  const float* on_g  = (const float*)d_in[7];
  const float* on_b  = (const float*)d_in[8];
  const float* W_q   = (const float*)d_in[9];
  const float* W_kv  = (const float*)d_in[10];
  const float* W_out = (const float*)d_in[11];
  const float* gamma = (const float*)d_in[12];
  const float* W1    = (const float*)d_in[13];
  const float* b1    = (const float*)d_in[14];
  const float* bn1_g = (const float*)d_in[15];
  const float* bn1_b = (const float*)d_in[16];
  const float* W2    = (const float*)d_in[17];
  const float* b2    = (const float*)d_in[18];
  const float* bn2_g = (const float*)d_in[19];
  const float* bn2_b = (const float*)d_in[20];
  float* yout = (float*)d_out;

  // Bump allocator over d_ws (deterministic layout every call).
  char* wsp = (char*)d_ws;
  size_t off = 0;
  auto alloc = [&](size_t bytes) -> void* {
    void* p = wsp + off;
    off = (off + bytes + 255) & ~(size_t)255;
    return p;
  };

  const long BCL = (long)Bc * CHc * Lc;
  bf16_t* cn_bf   = (bf16_t*)alloc(BCL * 2);
  bf16_t* qn_bf   = (bf16_t*)alloc(BCL * 2);
  bf16_t* Wq_bf   = (bf16_t*)alloc((long)CHc * CHc * 2);
  bf16_t* Wkv_bf  = (bf16_t*)alloc((long)2 * CHc * CHc * 2);
  bf16_t* Wout_bf = (bf16_t*)alloc((long)CHc * CHc * 2);
  bf16_t* W1_bf   = (bf16_t*)alloc((long)MLPc * CHc * 2);
  bf16_t* W2_bf   = (bf16_t*)alloc((long)CHc * MLPc * 2);
  float*  kv_f    = (float*)alloc((long)Bc * 2 * CHc * Lc * 4);
  float*  q_f     = (float*)alloc(BCL * 4);
  float*  qf_f    = (float*)alloc((long)BHc * Lc * DHc * 4);
  bf16_t* qf_b    = (bf16_t*)alloc((long)BHc * Lc * DHc * 2);
  float*  kf_f    = (float*)alloc((long)BHc * Lc * DHc * 4);
  float*  Qs_f    = (float*)alloc((long)BHc * TOPKc * DHc * 4);
  float*  mind_f  = (float*)alloc((long)BHc * Lc * 4);
  int*    idx_i   = (int*)alloc((long)BHc * TKPc * 4);
  bf16_t* kselT_b = (bf16_t*)alloc((long)BHc * DHc * TKPc * 2);
  bf16_t* vsel_b  = (bf16_t*)alloc((long)BHc * TKPc * DHc * 2);
  float*  scores  = (float*)alloc((long)Lc * TKPc * 4);   // per-head, L2-resident
  bf16_t* attn_b  = (bf16_t*)alloc((long)Lc * TKPc * 2);  // per-head
  float*  out4_f  = (float*)alloc((long)BHc * Lc * DHc * 4);
  bf16_t* proj_in = (bf16_t*)alloc(BCL * 2);
  float*  proj_f  = (float*)alloc(BCL * 4);
  float*  ao_f    = (float*)alloc(BCL * 4);
  bf16_t* ao_b    = (bf16_t*)alloc(BCL * 2);
  float*  h1_f    = (float*)alloc((long)Bc * MLPc * Lc * 4);
  bf16_t* h1_b    = (bf16_t*)alloc((long)Bc * MLPc * Lc * 2);
  float*  h2_f    = (float*)alloc(BCL * 4);

  auto blocks = [](long n) { return (int)((n + 255) / 256); };

  // 1) weights -> bf16
  f2bf_kernel<<<blocks((long)CHc * CHc), 256, 0, stream>>>(W_q, Wq_bf, (long)CHc * CHc);
  f2bf_kernel<<<blocks((long)2 * CHc * CHc), 256, 0, stream>>>(W_kv, Wkv_bf, (long)2 * CHc * CHc);
  f2bf_kernel<<<blocks((long)CHc * CHc), 256, 0, stream>>>(W_out, Wout_bf, (long)CHc * CHc);
  f2bf_kernel<<<blocks((long)MLPc * CHc), 256, 0, stream>>>(W1, W1_bf, (long)MLPc * CHc);
  f2bf_kernel<<<blocks((long)CHc * MLPc), 256, 0, stream>>>(W2, W2_bf, (long)CHc * MLPc);

  // 2) channel LayerNorms -> bf16 GEMM operands
  chan_ln_kernel<<<Bc * Lc, 256, 0, stream>>>(ctx,  cn_g, cn_b, cn_bf, (float*)nullptr);
  chan_ln_kernel<<<Bc * Lc, 256, 0, stream>>>(qsrc, qn_g, qn_b, qn_bf, (float*)nullptr);

  // 3) kv = W_kv x cn (512x4096, K=256, batched over B); 4) q = W_q x qn
  launch_gemm(Wkv_bf, 0, cn_bf, (long)CHc * Lc, kv_f, (long)2 * CHc * Lc,
              2 * CHc, Lc, CHc, Bc, stream);
  launch_gemm(Wq_bf, 0, qn_bf, (long)CHc * Lc, q_f, (long)CHc * Lc,
              CHc, Lc, CHc, Bc, stream);

  // 5) fold + l2norm
  fold_l2_kernel<<<blocks((long)BHc * Lc), 256, 0, stream>>>(q_f, kv_f, qf_f, qf_b, kf_f);

  // 6) Q_small gather, 7) L1 min-distance, 8) stable top-k select, 9) gather k/v
  gather_qs_kernel<<<blocks((long)BHc * TOPKc * DHc), 256, 0, stream>>>(qf_f, rind, Qs_f);
  {
    dim3 g(Lc / 256, BHc);
    l1_min_kernel<<<g, 256, 0, stream>>>(kf_f, Qs_f, mind_f);
  }
  select_topk_kernel<<<BHc, 256, 0, stream>>>(mind_f, idx_i);
  gather_kv_kernel<<<blocks((long)BHc * TKPc * DHc), 256, 0, stream>>>(
      idx_i, kf_f, kv_f, kselT_b, vsel_b);

  // 10) attention per (b,h): scores/attn stay L2-resident (9.4 MB << 192 MB L2)
  for (int bh = 0; bh < BHc; ++bh) {
    launch_gemm(qf_b + (long)bh * Lc * DHc, 0,
                kselT_b + (long)bh * DHc * TKPc, 0,
                scores, 0, Lc, TKPc, DHc, 1, stream);
    softmax_kernel<<<Lc / 8, 256, 0, stream>>>(scores, attn_b);
    launch_gemm(attn_b, 0,
                vsel_b + (long)bh * TKPc * DHc, 0,
                out4_f + (long)bh * Lc * DHc, 0, Lc, DHc, TKPc, 1, stream);
  }

  // 11) un-fold -> (B,C,L) bf16, 12) W_out projection
  unfold_kernel<<<blocks(BCL), 256, 0, stream>>>(out4_f, proj_in);
  launch_gemm(Wout_bf, 0, proj_in, (long)CHc * Lc, proj_f, (long)CHc * Lc,
              CHc, Lc, CHc, Bc, stream);

  // 13) out-LN + gamma residual
  ln_axpy_kernel<<<Bc * Lc, 256, 0, stream>>>(proj_f, on_g, on_b, gamma, qsrc, ao_f, ao_b);

  // 14-16) MLP
  launch_gemm(W1_bf, 0, ao_b, (long)CHc * Lc, h1_f, (long)MLPc * Lc,
              MLPc, Lc, CHc, Bc, stream);
  relu_bn_kernel<<<blocks((long)Bc * MLPc * Lc), 256, 0, stream>>>(
      h1_f, b1, bn1_g, bn1_b, h1_b);
  launch_gemm(W2_bf, 0, h1_b, (long)MLPc * Lc, h2_f, (long)CHc * Lc,
              CHc, Lc, MLPc, Bc, stream);

  // 17) final residual add -> output
  final_kernel<<<blocks(BCL), 256, 0, stream>>>(ao_f, h2_f, b2, bn2_g, bn2_b, yout);
}
